// AttnModel_13692355740200
// MI455X (gfx1250) — compile-verified, matrix operand order checked by reading, not run on hardware
//
#include <hip/hip_runtime.h>
#include <hip/hip_bf16.h>

typedef __attribute__((ext_vector_type(16))) _Float16 v16h;
typedef __attribute__((ext_vector_type(8)))  float    v8f;
typedef int v4i __attribute__((vector_size(16)));

#define BB   2048
#define NN   64
#define DD   128
#define DMM  256
#define HH   8
#define EE   6
#define KSTR 264   // padded LDS row stride in halves: 528B, 16B-aligned, bank-friendly

// ---------------------------------------------------------------------------
// CDNA5 async global->LDS helpers (guarded; fallback = direct 16B copy)
// builtin signature (from hipcc diagnostic): param0 = v4i addrspace(1)*,
// so pass typed int4 pointers in the right address spaces.
// ---------------------------------------------------------------------------
#if defined(__gfx1250__) && __has_builtin(__builtin_amdgcn_global_load_async_to_lds_b128)
#define HAVE_ASYNC_LDS 1
typedef __attribute__((address_space(1))) v4i* gv4p;
typedef __attribute__((address_space(3))) v4i* lv4p;
#endif

__device__ __forceinline__ void copy16_g2l(void* lds, const void* glob) {
#ifdef HAVE_ASYNC_LDS
    __builtin_amdgcn_global_load_async_to_lds_b128((gv4p)(glob), (lv4p)(lds), 0, 0);
#else
    *(float4*)lds = *(const float4*)glob;
#endif
}

__device__ __forceinline__ void wait_async0() {
#ifdef HAVE_ASYNC_LDS
#if __has_builtin(__builtin_amdgcn_s_wait_asynccnt)
    __builtin_amdgcn_s_wait_asynccnt(0);
#else
    asm volatile("s_wait_asynccnt 0x0" ::: "memory");
#endif
#endif
}

// ---------------------------------------------------------------------------
// f32 -> f16 weight conversion (weights stay L2-resident afterwards)
// ---------------------------------------------------------------------------
__global__ void cvt_f16(const float* __restrict__ x, _Float16* __restrict__ y, int n) {
    int i = blockIdx.x * 256 + threadIdx.x;
    if (i < n) y[i] = (_Float16)x[i];
}

// ---------------------------------------------------------------------------
// Per-batch q = [LN(src), src_t] -> f16  (q identical across N in reference)
// ---------------------------------------------------------------------------
__global__ void prep_q(const float* __restrict__ src, const float* __restrict__ src_t,
                       const float* __restrict__ g, const float* __restrict__ be,
                       _Float16* __restrict__ qh) {
    int b = blockIdx.x, t = threadIdx.x;       // 128 threads
    __shared__ float red[8], red2[8], mv[2];
    float x = src[(size_t)b * DD + t];
    float s = x, s2 = x * x;
    int l = t & 31, w = t >> 5;
    for (int off = 16; off; off >>= 1) {
        s  += __shfl_xor(s,  off, 32);
        s2 += __shfl_xor(s2, off, 32);
    }
    if (l == 0) { red[w] = s; red2[w] = s2; }
    __syncthreads();
    if (t == 0) {
        float a = 0.f, a2 = 0.f;
        for (int i = 0; i < 4; i++) { a += red[i]; a2 += red2[i]; }
        mv[0] = a / 128.f; mv[1] = a2 / 128.f;
    }
    __syncthreads();
    float mu = mv[0], var = mv[1] - mu * mu, rs = rsqrtf(var + 1e-5f);
    qh[(size_t)b * DMM + t]       = (_Float16)((x - mu) * rs * g[t] + be[t]);
    qh[(size_t)b * DMM + 128 + t] = (_Float16)src_t[(size_t)b * DD + t];
}

// ---------------------------------------------------------------------------
// Qe[b,e,:] = q_b @ Wq[e]^T  for all 6 edge types.
// grid = B/16 blocks, 256 threads (8 waves). WMMA 16x16x32 f16.
// ---------------------------------------------------------------------------
__global__ void qe_wmma(const _Float16* __restrict__ qh,
                        const _Float16* __restrict__ Wqh,
                        _Float16* __restrict__ Qe) {
    int bt = blockIdx.x;
    int t = threadIdx.x, w = t >> 5, l = t & 31;
    int r = l & 15, hl = l >> 4, base = hl * 8;
    int brow = bt * 16 + r;

    v16h afr[8];
    #pragma unroll
    for (int ks = 0; ks < 8; ks++) {
        const _Float16* p = qh + (size_t)brow * DMM + ks * 32 + base;
        #pragma unroll
        for (int i = 0; i < 8; i++) { afr[ks][i] = p[i]; afr[ks][8 + i] = p[16 + i]; }
    }

    #pragma unroll 1
    for (int tt = w; tt < 96; tt += 8) {          // 6 e-types x 16 n-tiles
        int e = tt / 16, nt = tt % 16;
        v8f c = {};
        #pragma unroll
        for (int ks = 0; ks < 8; ks++) {
            const _Float16* bp = Wqh + (size_t)e * 65536 + (size_t)(nt * 16 + r) * DMM + ks * 32 + base;
            v16h bf;
            #pragma unroll
            for (int i = 0; i < 8; i++) { bf[i] = bp[i]; bf[8 + i] = bp[16 + i]; }
            c = __builtin_amdgcn_wmma_f32_16x16x32_f16(false, afr[ks], false, bf,
                                                       (short)0, c, false, false);
        }
        #pragma unroll
        for (int v = 0; v < 8; v++) {
            int b2 = bt * 16 + v + 8 * hl;
            Qe[((size_t)(b2 * EE + e)) * DMM + nt * 16 + r] = (_Float16)c[v];
        }
    }
}

// ---------------------------------------------------------------------------
// Fused per-batch kernel.
// LDS map (dynamic):
//   kh    [0      , 33792 )  64 x KSTR f16   input tile [LN(seq+seq_e), seq_t]
//   Ksh   [33792  , 67584 )  64 x KSTR f16
//   Vsh   [67584  , 101376)  64 x KSTR f16
//   stage [101376 , 134144)  64x128 f32      async-staged seq_t
//   attw  [134144 , 136192)  8x64 f32
//   outv  [136192 , 137216)  256 f32
//   yv    [137216 , 138240)  256 f32
//   red   [138240 , 138496)
//   hsh   [138496 , 139008)  128 f32
// ---------------------------------------------------------------------------
__global__ void __launch_bounds__(256)
fused_attn(const float* __restrict__ seq, const float* __restrict__ seq_t,
           const float* __restrict__ seq_e,
           const float* __restrict__ ln2_g, const float* __restrict__ ln2_b,
           const _Float16* __restrict__ Wkh, const _Float16* __restrict__ Wvh,
           const _Float16* __restrict__ Qe,  const float* __restrict__ rp,
           const int* __restrict__ etype, const int* __restrict__ utype,
           const unsigned char* __restrict__ msk,
           const float* __restrict__ fc_w, const float* __restrict__ fc_b,
           const float* __restrict__ ln3_g, const float* __restrict__ ln3_b,
           const float* __restrict__ m1_w, const float* __restrict__ m1_b,
           const float* __restrict__ m2_w, const float* __restrict__ m2_b,
           const float* __restrict__ src,
           float* __restrict__ out_final, float* __restrict__ out_attn) {
    extern __shared__ char smem[];
    _Float16* kh    = (_Float16*)smem;
    _Float16* Ksh   = (_Float16*)(smem + 33792);
    _Float16* Vsh   = (_Float16*)(smem + 67584);
    float*    stage = (float*)(smem + 101376);
    float*    attw  = (float*)(smem + 134144);
    float*    outv  = (float*)(smem + 136192);
    float*    yv    = (float*)(smem + 137216);
    float*    red   = (float*)(smem + 138240);
    float*    hsh   = (float*)(smem + 138496);

    int b = blockIdx.x, t = threadIdx.x;
    int w = t >> 5, l = t & 31;
    int u = utype[b];

    // ---- P0: kick async seq_t tile copy (32KB) global -> LDS ------------
    {
        const float* gsrc = seq_t + (size_t)b * (NN * DD);
        #pragma unroll
        for (int k = 0; k < 8; k++) {
            int idx = t + k * 256;               // 2048 x 16B chunks
            copy16_g2l(stage + idx * 4, gsrc + idx * 4);
        }
    }

    // ---- P1: LN(seq+seq_e) -> kh[:,0:128] (overlapped with async copy) --
    if (t < NN) {
        const float* xs = seq   + ((size_t)b * NN + t) * DD;
        const float* xe = seq_e + ((size_t)b * NN + t) * DD;
        float s = 0.f, s2 = 0.f;
        for (int i = 0; i < DD; i++) { float x = xs[i] + xe[i]; s += x; s2 += x * x; }
        float mu = s / 128.f, var = s2 / 128.f - mu * mu, rs = rsqrtf(var + 1e-5f);
        for (int i = 0; i < DD; i++) {
            float x = xs[i] + xe[i];
            kh[t * KSTR + i] = (_Float16)((x - mu) * rs * ln2_g[i] + ln2_b[i]);
        }
    }
    wait_async0();
    __syncthreads();
    for (int idx = t; idx < NN * DD; idx += 256) {
        int n = idx >> 7, i = idx & 127;
        kh[n * KSTR + 128 + i] = (_Float16)stage[idx];
    }
    __syncthreads();

    // ---- P2: K (waves 0-3) / V (waves 4-7), B frags reused over 4 M-tiles
    {
        int r = l & 15, hl = l >> 4, base = hl * 8;
        int isV = w >> 2, ng = w & 3;            // wave owns 4 n-tiles
        const _Float16* Wm  = isV ? Wvh : Wkh;
        _Float16*       Osh = isV ? Vsh : Ksh;
        int et4[4];
        #pragma unroll
        for (int mt = 0; mt < 4; mt++) et4[mt] = etype[(size_t)b * NN + mt * 16 + r];
        const v16h zero = {};

        #pragma unroll 1
        for (int nq = 0; nq < 4; nq++) {
            int nt = ng * 4 + nq;
            v8f c[4] = {{}, {}, {}, {}};
            #pragma unroll 1
            for (int e = 0; e < EE; e++) {
                // B fragments for (e, nt): loaded once, feed up to 32 wmma
                v16h bf[8];
                const _Float16* wb = Wm + (size_t)e * 65536 + (size_t)(nt * 16 + r) * DMM + base;
                #pragma unroll
                for (int ks = 0; ks < 8; ks++) {
                    const _Float16* bp = wb + ks * 32;
                    #pragma unroll
                    for (int i = 0; i < 8; i++) { bf[ks][i] = bp[i]; bf[ks][8 + i] = bp[16 + i]; }
                }
                #pragma unroll
                for (int mt = 0; mt < 4; mt++) {
                    bool on = (et4[mt] == e);
                    if (__ballot(on) == 0) continue;   // wave-uniform: whole tile zero
                    const _Float16* ap = kh + (mt * 16 + r) * KSTR + base;
                    #pragma unroll
                    for (int ks = 0; ks < 8; ks++) {
                        v16h af;
                        const _Float16* p = ap + ks * 32;
                        #pragma unroll
                        for (int i = 0; i < 8; i++) { af[i] = p[i]; af[8 + i] = p[16 + i]; }
                        v16h am = on ? af : zero;
                        c[mt] = __builtin_amdgcn_wmma_f32_16x16x32_f16(false, am, false, bf[ks],
                                                                       (short)0, c[mt], false, false);
                    }
                }
            }
            #pragma unroll
            for (int mt = 0; mt < 4; mt++) {
                #pragma unroll
                for (int v = 0; v < 8; v++)
                    Osh[(mt * 16 + v + 8 * hl) * KSTR + nt * 16 + r] = (_Float16)c[mt][v];
            }
        }
    }
    __syncthreads();

    // ---- P3: per-head scores + wave32 softmax ---------------------------
    {
        int h = w;
        float sc0, sc1;
        #pragma unroll
        for (int q2 = 0; q2 < 2; q2++) {
            int n = l + q2 * 32;
            int et = etype[(size_t)b * NN + n];
            const _Float16* qp = Qe + ((size_t)(b * EE + et)) * DMM + h * 32;
            const _Float16* kp = Ksh + n * KSTR + h * 32;
            float s = 0.f;
            #pragma unroll
            for (int j = 0; j < 32; j++) s += (float)qp[j] * (float)kp[j];
            s *= rp[u * EE + et] * 0.17677669529663687f;   // pri / sqrt(32)
            if (msk[(size_t)b * NN + n]) s = -1e10f;
            if (q2 == 0) sc0 = s; else sc1 = s;
        }
        float m = fmaxf(sc0, sc1);
        for (int off = 16; off; off >>= 1) m = fmaxf(m, __shfl_xor(m, off, 32));
        float e0 = __expf(sc0 - m), e1 = __expf(sc1 - m);
        float sum = e0 + e1;
        for (int off = 16; off; off >>= 1) sum += __shfl_xor(sum, off, 32);
        float inv = 1.f / sum;
        float w0 = e0 * inv, w1 = e1 * inv;
        out_attn[((size_t)h * BB + b) * NN + l]      = w0;
        out_attn[((size_t)h * BB + b) * NN + l + 32] = w1;
        attw[h * NN + l] = w0; attw[h * NN + l + 32] = w1;
    }
    __syncthreads();

    // ---- P4: out[h*32+j] = sum_n attw[h,n] * V[n, h*32+j] ---------------
    {
        int h = t >> 5, j = t & 31;
        float acc = 0.f;
        for (int n = 0; n < NN; n++)
            acc += attw[h * NN + n] * (float)Vsh[n * KSTR + h * 32 + j];
        outv[t] = acc;
    }
    __syncthreads();

    // ---- P5: gathered fc + LN over 256 ----------------------------------
    float y = fc_b[u * DMM + t];
    {
        const float* fw = fc_w + (size_t)u * 65536 + (size_t)t * DMM;
        for (int i = 0; i < DMM; i++) y += fw[i] * outv[i];
    }
    {
        float s = y, s2 = y * y;
        for (int off = 16; off; off >>= 1) {
            s  += __shfl_xor(s,  off, 32);
            s2 += __shfl_xor(s2, off, 32);
        }
        if (l == 0) { red[w] = s; red[8 + w] = s2; }
        __syncthreads();
        if (t == 0) {
            float a = 0.f, a2 = 0.f;
            for (int i = 0; i < 8; i++) { a += red[i]; a2 += red[8 + i]; }
            red[16] = a / 256.f; red[17] = a2 / 256.f;
        }
        __syncthreads();
        float mu = red[16], var = red[17] - mu * mu, rs = rsqrtf(var + 1e-5f);
        yv[t] = (y - mu) * rs * ln3_g[t] + ln3_b[t];
    }
    __syncthreads();

    // ---- P6: m1 (128 x 384) + relu --------------------------------------
    if (t < 128) {
        float acc = m1_b[t];
        const float* mw = m1_w + (size_t)t * 384;
        for (int i = 0; i < 256; i++) acc += mw[i] * yv[i];
        const float* sp = src + (size_t)b * DD;
        for (int i = 0; i < 128; i++) acc += mw[256 + i] * sp[i];
        hsh[t] = fmaxf(acc, 0.f);
    }
    __syncthreads();

    // ---- P7: m2 (128 x 128) -> final ------------------------------------
    if (t < 128) {
        float acc = m2_b[t];
        const float* mw = m2_w + (size_t)t * 128;
        for (int i = 0; i < 128; i++) acc += mw[i] * hsh[i];
        out_final[(size_t)b * DD + t] = acc;
    }
}

// ---------------------------------------------------------------------------
extern "C" void kernel_launch(void* const* d_in, const int* in_sizes, int n_in,
                              void* d_out, int out_size, void* d_ws, size_t ws_size,
                              hipStream_t stream) {
    const float* src    = (const float*)d_in[0];
    const float* src_t  = (const float*)d_in[1];
    const float* seq    = (const float*)d_in[2];
    const float* seq_t  = (const float*)d_in[3];
    const float* seq_e  = (const float*)d_in[4];
    const float* Wq     = (const float*)d_in[5];
    const float* Wk     = (const float*)d_in[6];
    const float* Wv     = (const float*)d_in[7];
    const float* rp     = (const float*)d_in[8];
    const float* fc_w   = (const float*)d_in[9];
    const float* fc_b   = (const float*)d_in[10];
    const float* ln1_g  = (const float*)d_in[11];
    const float* ln1_b  = (const float*)d_in[12];
    const float* ln2_g  = (const float*)d_in[13];
    const float* ln2_b  = (const float*)d_in[14];
    const float* ln3_g  = (const float*)d_in[15];
    const float* ln3_b  = (const float*)d_in[16];
    const float* m1_w   = (const float*)d_in[17];
    const float* m1_b   = (const float*)d_in[18];
    const float* m2_w   = (const float*)d_in[19];
    const float* m2_b   = (const float*)d_in[20];
    const int*   etype  = (const int*)d_in[21];
    const int*   utype  = (const int*)d_in[22];
    // d_in[23] = seq_vtype (unused by reference math)
    const unsigned char* mask = (const unsigned char*)d_in[24];

    char* ws = (char*)d_ws;
    _Float16* qh  = (_Float16*)ws;                   // B*256 f16      = 1,048,576 B
    _Float16* Qe  = (_Float16*)(ws + 1048576);       // B*6*256 f16    = 6,291,456 B
    _Float16* Wqh = (_Float16*)(ws + 7340032);       // 6*256*256 f16  =   786,432 B
    _Float16* Wkh = (_Float16*)(ws + 8126464);
    _Float16* Wvh = (_Float16*)(ws + 8912896);       // ends ~9.7 MB

    float* out_final = (float*)d_out;
    float* out_attn  = out_final + (size_t)BB * DD;

    int wn = EE * DMM * DMM;
    cvt_f16<<<(wn + 255) / 256, 256, 0, stream>>>(Wq, Wqh, wn);
    cvt_f16<<<(wn + 255) / 256, 256, 0, stream>>>(Wk, Wkh, wn);
    cvt_f16<<<(wn + 255) / 256, 256, 0, stream>>>(Wv, Wvh, wn);
    prep_q<<<BB, 128, 0, stream>>>(src, src_t, ln1_g, ln1_b, qh);
    qe_wmma<<<BB / 16, 256, 0, stream>>>(qh, Wqh, Qe);
    fused_attn<<<BB, 256, 139264, stream>>>(seq, seq_t, seq_e, ln2_g, ln2_b,
                                            Wkh, Wvh, Qe, rp, etype, utype, mask,
                                            fc_w, fc_b, ln3_g, ln3_b,
                                            m1_w, m1_b, m2_w, m2_b, src,
                                            out_final, out_attn);
}
